// attentiona_36266703848143
// MI455X (gfx1250) — matrix operand order repaired and checked by reading, not compiled, not run on hardware
//
#include <hip/hip_runtime.h>
#include <hip/hip_bf16.h>

#ifndef __has_builtin
#define __has_builtin(x) 0
#endif

#if __has_builtin(__builtin_amdgcn_global_load_async_to_lds_b128) && \
    __has_builtin(__builtin_amdgcn_s_wait_asynccnt)
#define USE_ASYNC_LDS 1
#else
#define USE_ASYNC_LDS 0
#endif

typedef __attribute__((ext_vector_type(16))) _Float16 v16h;
typedef __attribute__((ext_vector_type(8)))  _Float16 v8h;
typedef __attribute__((ext_vector_type(8)))  float    v8f;
typedef int v4i __attribute__((vector_size(16)));   // matches builtin's pointee type

constexpr int B_ = 2, C_ = 2048, D_ = 1024, H_ = 16, HD_ = 64;
constexpr int DQKV = 3 * D_;   // 3072 (q | k | v)
constexpr int M_ = B_ * C_;    // 4096 rows total

static __device__ __forceinline__ v16h cat16(v8h lo, v8h hi) {
  return __builtin_shufflevector(lo, hi, 0,1,2,3,4,5,6,7,8,9,10,11,12,13,14,15);
}
static __device__ __forceinline__ v8h ld8(const _Float16* p) {   // 16B LDS/global load
  return __builtin_bit_cast(v8h, *(const uint4*)p);
}
static __device__ __forceinline__ v16h ldfrag(const _Float16* lo, const _Float16* hi) {
  return cat16(ld8(lo), ld8(hi));
}
static __device__ __forceinline__ v8f wmma_f16(v16h a, v16h b, v8f c) {
  // D = A(16x32 f16) x B(32x16 f16) + C(16x16 f32)
  return __builtin_amdgcn_wmma_f32_16x16x32_f16(false, a, false, b, (short)0, c, false, false);
}

// gfx1250 async global->LDS (ASYNCcnt-tracked); sync fallback keeps compile safe.
static __device__ __forceinline__ void async_cp16(void* l, const void* g) {
#if USE_ASYNC_LDS
  __builtin_amdgcn_global_load_async_to_lds_b128(
      (__attribute__((address_space(1))) v4i*)g,
      (__attribute__((address_space(3))) v4i*)l, 0, 0);
#else
  *(uint4*)l = *(const uint4*)g;
#endif
}
template<int N>
static __device__ __forceinline__ void wait_async() {
#if USE_ASYNC_LDS
  __builtin_amdgcn_s_wait_asynccnt(N);
#endif
}

// ---------------------------------------------------------------- f32 -> f16
__global__ __launch_bounds__(256) void cvt_f32_f16(const float* __restrict__ s,
                                                   _Float16* __restrict__ d, int n) {
  int i = (blockIdx.x * 256 + threadIdx.x) * 4;
  if (i + 3 < n) {
    float4 f = *(const float4*)(s + i);
    d[i]   = (_Float16)f.x; d[i+1] = (_Float16)f.y;
    d[i+2] = (_Float16)f.z; d[i+3] = (_Float16)f.w;
  }
}

// ---------------------------------------------------------------- LayerNorm
__global__ __launch_bounds__(256) void ln_kernel(const float* __restrict__ x,
                                                 const float* __restrict__ g,
                                                 const float* __restrict__ bta,
                                                 _Float16* __restrict__ xn) {
  __shared__ float red[256];
  const int row = blockIdx.x, tid = threadIdx.x;
  const float* xr = x + (size_t)row * D_;
  float v[4], s = 0.f, s2 = 0.f;
  for (int i = 0; i < 4; ++i) { float t = xr[tid + i*256]; v[i] = t; s += t; s2 += t*t; }
  red[tid] = s; __syncthreads();
  for (int st = 128; st > 0; st >>= 1) { if (tid < st) red[tid] += red[tid+st]; __syncthreads(); }
  const float mu = red[0] * (1.0f / D_);
  __syncthreads();
  red[tid] = s2; __syncthreads();
  for (int st = 128; st > 0; st >>= 1) { if (tid < st) red[tid] += red[tid+st]; __syncthreads(); }
  const float var  = red[0] * (1.0f / D_) - mu * mu;
  const float rstd = rsqrtf(var + 1e-5f);
  _Float16* orow = xn + (size_t)row * D_;
  for (int i = 0; i < 4; ++i) {
    int c = tid + i*256;
    orow[c] = (_Float16)((v[i] - mu) * rstd * g[c] + bta[c]);
  }
}

// ------------------------------------------- Out[M,N] = A[M,K] * W[N,K]^T (NT)
// Double-buffered LDS pipeline driven by async global->LDS DMA.
template<bool F16OUT>
__global__ __launch_bounds__(256) void gemm_nt_kernel(
    const _Float16* __restrict__ A, const _Float16* __restrict__ W,
    void* __restrict__ Out, int M, int N, int K, int ldout) {
  __shared__ _Float16 As[2][128][40];   // 128 rows x 32 K-halves (+8 pad), x2 buffers
  __shared__ _Float16 Bs[2][128][40];
  const int tid = threadIdx.x;
  const int w = tid >> 5, lane = tid & 31;
  const int lrow = lane & 15, kh = lane >> 4;
  const int wm = (w & 1) * 64, wn = (w >> 1) * 32;
  const int m0 = blockIdx.x * 128, n0 = blockIdx.y * 128;
  const int lr = tid >> 1, seg = (tid & 1) * 16;
  const _Float16* arow = A + (size_t)(m0 + lr) * K + seg;
  const _Float16* wrow = W + (size_t)(n0 + lr) * K + seg;

  auto stage = [&](int buf, int k0) {   // 4 x b128 async per thread
    async_cp16(&As[buf][lr][seg],     arow + k0);
    async_cp16(&As[buf][lr][seg + 8], arow + k0 + 8);
    async_cp16(&Bs[buf][lr][seg],     wrow + k0);
    async_cp16(&Bs[buf][lr][seg + 8], wrow + k0 + 8);
  };

  v8f acc[4][2] = {};
  const int nk = K / 32;
  stage(0, 0);
  for (int ki = 0; ki < nk; ++ki) {
    const int cur = ki & 1;
    if (ki + 1 < nk) { stage(cur ^ 1, (ki + 1) * 32); wait_async<4>(); }
    else             { wait_async<0>(); }
    __syncthreads();                    // tile ki fully in LDS for all waves
    v16h bf[2];
    for (int nt = 0; nt < 2; ++nt) {
      const int col = wn + nt*16 + lrow;            // B-frag: lane=N col, K contiguous
      bf[nt] = ldfrag(&Bs[cur][col][kh*16], &Bs[cur][col][kh*16 + 8]);
    }
    for (int mt = 0; mt < 4; ++mt) {
      const int row = wm + mt*16 + lrow;            // A-frag: lane=M row, K contiguous
      v16h af = ldfrag(&As[cur][row][kh*16], &As[cur][row][kh*16 + 8]);
      acc[mt][0] = wmma_f16(af, bf[0], acc[mt][0]);
      acc[mt][1] = wmma_f16(af, bf[1], acc[mt][1]);
    }
    __syncthreads();                    // reads done before buffer reuse
  }
  for (int mt = 0; mt < 4; ++mt)
    for (int nt = 0; nt < 2; ++nt)
      for (int r = 0; r < 8; ++r) {
        const int grow = m0 + wm + mt*16 + r + kh*8;   // C-layout: VGPR r -> M=r / r+8
        const int gcol = n0 + wn + nt*16 + lrow;
        const float v = acc[mt][nt][r];
        if constexpr (F16OUT) ((_Float16*)Out)[(size_t)grow * ldout + gcol] = (_Float16)v;
        else                  ((float*)Out)[(size_t)grow * ldout + gcol] = v;
      }
}

// ---------------------------------------------------------------- RoPE (q,k)
__global__ __launch_bounds__(256) void rope_kernel(_Float16* __restrict__ qkv,
                                                   const float* __restrict__ theta_p) {
  const int row = blockIdx.x;           // b*C + pos
  const int pos = row & (C_ - 1);
  const float thf = theta_p[0] * (1.0f / 220.0f);
  for (int it = 0; it < 4; ++it) {
    const int j = it * 256 + threadIdx.x;   // {q,k} x 16 heads x 32 pairs
    const int sel = j >> 9, rem = j & 511;
    const int head = rem >> 5, i = rem & 31;
    // freqs_base[i] = 0.2 * (21^(i/31) - 1)
    const float fb = 0.2f * (__powf(21.0f, (float)i * (1.0f/31.0f)) - 1.0f);
    const float ang = (float)pos * thf * fb;
    float sn, cs; __sincosf(ang, &sn, &cs);
    const size_t base = (size_t)row * DQKV + sel * D_ + head * HD_ + 2*i;
    const float xe = (float)qkv[base], xo = (float)qkv[base + 1];
    qkv[base]     = (_Float16)(xe*cs - xo*sn);
    qkv[base + 1] = (_Float16)(xe*sn + xo*cs);
  }
}

// ---------------------------------------------------------------- attention
struct AttnSmem {
  _Float16 Vt[64][136];                 // V transposed: [d][key]
  float    scs[128];                    // per-key scale
  union {
    _Float16 Ksm[128][72];              // K block, row-major [key][d]
    _Float16 Pt[8][16][136];            // per-wave num tile [row][key]
  } u;
};

__global__ __launch_bounds__(256, 4) void attn_kernel(
    const _Float16* __restrict__ qkv, _Float16* __restrict__ wv,
    const int* __restrict__ use_mask_p) {
  __shared__ AttnSmem sm;
  const int tid = threadIdx.x, w = tid >> 5, lane = tid & 31;
  const int lrow = lane & 15, kh = lane >> 4;
  const int b = (int)blockIdx.y >> 4, h = (int)blockIdx.y & 15;
  const int q0 = (int)blockIdx.x * 128 + w * 16;    // this wave's 16 query rows
  const int um = use_mask_p[0];

  // Q fragments in registers for the whole key loop (K=HD=64 -> 2 k-steps)
  const _Float16* qrowp = qkv + (size_t)(b * C_ + q0 + lrow) * DQKV + h * HD_;
  v16h aq[2];
  for (int ks = 0; ks < 2; ++ks) {
    const int d0 = ks*32 + kh*16;
    aq[ks] = ldfrag(qrowp + d0, qrowp + d0 + 8);
  }
  v8f o[4] = {};
  float z[8] = {};

  const int lr = tid >> 1, seg = (tid & 1) * 32;
  for (int kb = 0; kb < C_ / 128; ++kb) {
    __syncthreads();                               // prior iter's Pt/Vt reads done
    {   // stage K via async DMA; V transposed + sc by hand
      const _Float16* kp = qkv + (size_t)(b*C_ + kb*128 + lr)*DQKV +   D_ + h*HD_ + seg;
      const _Float16* vp = qkv + (size_t)(b*C_ + kb*128 + lr)*DQKV + 2*D_ + h*HD_ + seg;
      for (int j = 0; j < 4; ++j)
        async_cp16(&sm.u.Ksm[lr][seg + j*8], kp + j*8);
      if (seg == 0) {
        const float k0v = (float)kp[0];
        sm.scs[lr] = (k0v == 0.0f) ? 1e-5f : 1.0f;
      }
      for (int j = 0; j < 32; ++j)                 // transpose scatter (overlaps DMA)
        sm.Vt[seg + j][lr] = vp[j];
      wait_async<0>();
    }
    __syncthreads();
    // S = Q K^T  (8 key sub-tiles of 16)
    v8f sAcc[8] = {};
    for (int ks = 0; ks < 2; ++ks)
      for (int t = 0; t < 8; ++t) {
        const int col = t*16 + lrow;               // B-frag: lane = key col
        v16h bk = ldfrag(&sm.u.Ksm[col][ks*32 + kh*16],
                         &sm.u.Ksm[col][ks*32 + kh*16 + 8]);
        sAcc[t] = wmma_f16(aq[ks], bk, sAcc[t]);
      }
    float scl[8];
    for (int t = 0; t < 8; ++t) scl[t] = sm.scs[t*16 + lrow];
    __syncthreads();                               // Ksm reads done before Pt write
    // transform -> num = 1 + p + p^2/2, row-sum partials, stage P as f16
    float zp[8] = {};
    for (int t = 0; t < 8; ++t) {
      const float sc = scl[t];
      const int key = kb*128 + t*16 + lrow;
      for (int r = 0; r < 8; ++r) {
        const int qrow = q0 + r + kh*8;            // C-layout row mapping
        const float x = sAcc[t][r] * 0.125f;       // HD^-0.5
        float p;
        if (um) {
          const bool msk = key > qrow;             // masked -> sigmoid(-inf)=0
          p = msk ? 0.0f : sc / (1.0f + __expf(-x * sc));
        } else {
          p = x * sc;
        }
        const float num = fmaf(0.5f * p, p, 1.0f + p);
        zp[r] += num;
        sm.u.Pt[w][r + kh*8][t*16 + lrow] = (_Float16)num;
      }
    }
    for (int r = 0; r < 8; ++r) {                  // sum cols across 16 lanes
      float tot = zp[r];
      tot += __shfl_xor(tot, 1, 16);
      tot += __shfl_xor(tot, 2, 16);
      tot += __shfl_xor(tot, 4, 16);
      tot += __shfl_xor(tot, 8, 16);
      z[r] += tot;
    }
    __syncthreads();
    // O += P V   (K = 128 keys -> 4 k-steps; N = HD = 64 -> 4 n-tiles)
    for (int k2 = 0; k2 < 4; ++k2) {
      v16h ap = ldfrag(&sm.u.Pt[w][lrow][k2*32 + kh*16],
                       &sm.u.Pt[w][lrow][k2*32 + kh*16 + 8]);
      for (int nt = 0; nt < 4; ++nt) {
        const int d = nt*16 + lrow;                // B-frag: lane = d col, K=key contiguous
        v16h bv = ldfrag(&sm.Vt[d][k2*32 + kh*16],
                         &sm.Vt[d][k2*32 + kh*16 + 8]);
        o[nt] = wmma_f16(ap, bv, o[nt]);
      }
    }
  }
  // normalize by Z (lane-half row mapping matches C-layout exactly) and store
  for (int nt = 0; nt < 4; ++nt)
    for (int r = 0; r < 8; ++r) {
      const int row = q0 + r + kh*8;
      const int col = h*HD_ + nt*16 + lrow;
      wv[(size_t)(b*C_ + row) * D_ + col] = (_Float16)(o[nt][r] / z[r]);
    }
}

// ---------------------------------------------------------------- launcher
extern "C" void kernel_launch(void* const* d_in, const int* in_sizes, int n_in,
                              void* d_out, int out_size, void* d_ws, size_t ws_size,
                              hipStream_t stream) {
  const float* x     = (const float*)d_in[0];
  const float* ln_w  = (const float*)d_in[1];
  const float* ln_b  = (const float*)d_in[2];
  const float* Wq    = (const float*)d_in[3];
  const float* Wkv   = (const float*)d_in[4];
  const float* Wout  = (const float*)d_in[5];
  const float* theta = (const float*)d_in[6];
  const int* use_mask = (const int*)d_in[7];

  char* ws = (char*)d_ws;
  _Float16* wh_qkv = (_Float16*)(ws);                  // [3072,1024] f16 (Wq;Wkv)
  _Float16* wh_out = (_Float16*)(ws + 6291456);        // [1024,1024] f16
  _Float16* xn_h   = (_Float16*)(ws + 8388608);        // [4096,1024] f16
  _Float16* qkv_h  = (_Float16*)(ws + 16777216);       // [4096,3072] f16
  _Float16* wv_h   = (_Float16*)(ws + 41943040);       // [4096,1024] f16

  cvt_f32_f16<<<1048576/1024, 256, 0, stream>>>(Wq,  wh_qkv,           1048576);
  cvt_f32_f16<<<2097152/1024, 256, 0, stream>>>(Wkv, wh_qkv + 1048576, 2097152);
  cvt_f32_f16<<<1048576/1024, 256, 0, stream>>>(Wout, wh_out,          1048576);

  ln_kernel<<<M_, 256, 0, stream>>>(x, ln_w, ln_b, xn_h);

  gemm_nt_kernel<true><<<dim3(M_/128, DQKV/128), 256, 0, stream>>>(
      xn_h, wh_qkv, qkv_h, M_, DQKV, D_, DQKV);

  rope_kernel<<<M_, 256, 0, stream>>>(qkv_h, theta);

  attn_kernel<<<dim3(C_/128, B_*H_), 256, 0, stream>>>(qkv_h, wv_h, use_mask);

  gemm_nt_kernel<false><<<dim3(M_/128, D_/128), 256, 0, stream>>>(
      wv_h, wh_out, d_out, M_, D_, D_, D_);
}
